// InformerModel_79182017069661
// MI455X (gfx1250) — matrix operand order repaired and checked by reading, not compiled, not run on hardware
//
#include <hip/hip_runtime.h>
#include <hip/hip_bf16.h>

typedef __attribute__((ext_vector_type(16))) _Float16 v16h;
typedef __attribute__((ext_vector_type(8)))  _Float16 v8h_t;
typedef __attribute__((ext_vector_type(8)))  float    v8f;

#define BM 128
#define BN 64
#define BK 32
#define LDT 40   // LDS row stride in halves (80B) -> conflict-free b128 reads

union FragU { v16h v; v8h_t h[2]; };

// C[M,N] = alpha * A[M,K] @ (TRANSB ? B[N,K]^T : B[K,N]) + bias, optional ReLU.
// Block = 128 threads (4 waves); block tile 128x64; each wave owns a 32x64 strip.
// blockIdx.z decomposed as (batch = z>>3, head = z&7) with separate strides.
template<bool TRANSB, bool RELU>
__global__ __launch_bounds__(128) void gemm_wmma_f16(
    const float* __restrict__ A, const float* __restrict__ B,
    float* __restrict__ C, const float* __restrict__ bias,
    int K, int lda, int ldb, int ldc,
    long long aSB, long long aSH, long long bSB, long long bSH,
    long long cSB, long long cSH, float alpha)
{
  __shared__ __align__(16) _Float16 As[BM * LDT];
  __shared__ __align__(16) _Float16 Bs[BN * LDT];

  const int zb = blockIdx.z >> 3, zh = blockIdx.z & 7;
  A += (long long)zb * aSB + (long long)zh * aSH;
  B += (long long)zb * bSB + (long long)zh * bSH;
  C += (long long)zb * cSB + (long long)zh * cSH;

  const int tid   = threadIdx.x;
  const int tileM = blockIdx.x * BM;
  const int tileN = blockIdx.y * BN;
  const int lane  = tid & 31;
  const int w     = tid >> 5;   // wave 0..3 -> 32-row strip of C tile
  const int g     = lane >> 4;  // half-wave group
  const int lr    = lane & 15;

  v8f acc[2][4];
  const v8f vzero = {0.f, 0.f, 0.f, 0.f, 0.f, 0.f, 0.f, 0.f};
#pragma unroll
  for (int s = 0; s < 2; ++s)
#pragma unroll
    for (int j = 0; j < 4; ++j) acc[s][j] = vzero;

  for (int kb = 0; kb < K; kb += BK) {
    // ---- stage A tile [128][32]: issue all loads, then convert f32->f16
    {
      const int r = tid >> 3;        // 0..15
      const int c = (tid & 7) << 2;  // 0..28
      float4 fa[8];
#pragma unroll
      for (int p = 0; p < 8; ++p)
        fa[p] = *(const float4*)(A + (long long)(tileM + r + p * 16) * lda + kb + c);
#pragma unroll
      for (int p = 0; p < 8; ++p) {
        _Float16* d = &As[(r + p * 16) * LDT + c];
        d[0] = (_Float16)fa[p].x; d[1] = (_Float16)fa[p].y;
        d[2] = (_Float16)fa[p].z; d[3] = (_Float16)fa[p].w;
      }
    }
    // ---- stage B tile as Bt[n][k], [64][32]
    if (TRANSB) {
      // B is [N,K] row-major: direct copy
      const int r = tid >> 3;
      const int c = (tid & 7) << 2;
      float4 fb[4];
#pragma unroll
      for (int p = 0; p < 4; ++p)
        fb[p] = *(const float4*)(B + (long long)(tileN + r + p * 16) * ldb + kb + c);
#pragma unroll
      for (int p = 0; p < 4; ++p) {
        _Float16* d = &Bs[(r + p * 16) * LDT + c];
        d[0] = (_Float16)fb[p].x; d[1] = (_Float16)fb[p].y;
        d[2] = (_Float16)fb[p].z; d[3] = (_Float16)fb[p].w;
      }
    } else {
      // B is [K,N] row-major: transpose while staging
      const int r = tid >> 4;          // k row 0..7
      const int c = (tid & 15) << 2;   // n 0..60
      float4 fb[4];
#pragma unroll
      for (int p = 0; p < 4; ++p)
        fb[p] = *(const float4*)(B + (long long)(kb + r + p * 8) * ldb + tileN + c);
#pragma unroll
      for (int p = 0; p < 4; ++p) {
        const int kr = r + p * 8;
        Bs[(c + 0) * LDT + kr] = (_Float16)fb[p].x;
        Bs[(c + 1) * LDT + kr] = (_Float16)fb[p].y;
        Bs[(c + 2) * LDT + kr] = (_Float16)fb[p].z;
        Bs[(c + 3) * LDT + kr] = (_Float16)fb[p].w;
      }
    }
    __syncthreads();

    // ---- A fragments for the wave's two 16-row strips
    // lane = row (strip + lr); K halves {8g..8g+7, 16+8g..+7}
    FragU af[2];
#pragma unroll
    for (int s = 0; s < 2; ++s) {
      const _Float16* ar = &As[(w * 32 + s * 16 + lr) * LDT];
      af[s].h[0] = *(const v8h_t*)(ar + 8 * g);
      af[s].h[1] = *(const v8h_t*)(ar + 16 + 8 * g);
    }

#pragma unroll
    for (int j = 0; j < 4; ++j) {
      // B fragment: lane = column (j*16+lr); K halves {8g..8g+7, 16+8g..+7}
      FragU bf;
      const _Float16* br = &Bs[(j * 16 + lr) * LDT + 8 * g];
      bf.h[0] = *(const v8h_t*)(br);
      bf.h[1] = *(const v8h_t*)(br + 16);
      acc[0][j] = __builtin_amdgcn_wmma_f32_16x16x32_f16(
          false, af[0].v, false, bf.v, (short)0, acc[0][j], false, false);
      acc[1][j] = __builtin_amdgcn_wmma_f32_16x16x32_f16(
          false, af[1].v, false, bf.v, (short)0, acc[1][j], false, false);
    }
    __syncthreads();
  }

  // ---- epilogue: C layout: VGPR t -> M = t + 8g, N = lr
#pragma unroll
  for (int s = 0; s < 2; ++s) {
#pragma unroll
    for (int j = 0; j < 4; ++j) {
      const int col = tileN + j * 16 + lr;
      const float bv = bias ? bias[col] : 0.0f;
#pragma unroll
      for (int t = 0; t < 8; ++t) {
        const int row = tileM + w * 32 + s * 16 + t + 8 * g;
        float r = alpha * acc[s][j][t] + bv;
        if (RELU) r = fmaxf(r, 0.0f);
        C[(long long)row * ldc + col] = r;
      }
    }
  }
}

// wave-per-row: exact top-u threshold (u-th largest incl. duplicates),
// mask scores < kth to -10000, stable softmax in place. L == 1024.
__global__ __launch_bounds__(256) void topk_softmax_kernel(float* __restrict__ S, int u)
{
  const int lane = threadIdx.x & 31;
  const int row  = blockIdx.x * 8 + (threadIdx.x >> 5);
  float* p = S + (long long)row * 1024;

  float vals[32], cur[32];
#pragma unroll
  for (int i = 0; i < 32; ++i) { vals[i] = p[lane + (i << 5)]; cur[i] = vals[i]; }

  const float NEG = -3.0e38f;
  float rowmax = NEG, kth = NEG;
  for (int it = 0; it < u; ++it) {
    float lm = NEG;
#pragma unroll
    for (int i = 0; i < 32; ++i) lm = fmaxf(lm, cur[i]);
    float wm = lm;
#pragma unroll
    for (int off = 16; off > 0; off >>= 1) wm = fmaxf(wm, __shfl_xor(wm, off, 32));
    if (it == 0) rowmax = wm;
    kth = wm;
    unsigned long long bal = __ballot(lm == wm);
    int owner = __ffsll((long long)bal) - 1;
    if (lane == owner) {
      bool done = false;
#pragma unroll
      for (int i = 0; i < 32; ++i)
        if (!done && cur[i] == wm) { cur[i] = NEG; done = true; }
    }
  }

  float sum = 0.f;
#pragma unroll
  for (int i = 0; i < 32; ++i) {
    float s = (vals[i] < kth) ? -10000.0f : vals[i];
    float e = __expf(s - rowmax);
    vals[i] = e;
    sum += e;
  }
#pragma unroll
  for (int off = 16; off > 0; off >>= 1) sum += __shfl_xor(sum, off, 32);
  const float inv = 1.0f / sum;
#pragma unroll
  for (int i = 0; i < 32; ++i) p[lane + (i << 5)] = vals[i] * inv;
}

// wave-per-row (d=512): h = LayerNorm(h + a) * g + b  (var = E[(x-mean)^2], eps 1e-5)
__global__ __launch_bounds__(256) void add_layernorm_kernel(
    float* __restrict__ h, const float* __restrict__ a,
    const float* __restrict__ gw, const float* __restrict__ bw)
{
  const int lane = threadIdx.x & 31;
  const int row  = blockIdx.x * 8 + (threadIdx.x >> 5);
  float* hp = h + (long long)row * 512;
  const float* ap = a + (long long)row * 512;

  float x[16];
  float s = 0.f;
#pragma unroll
  for (int i = 0; i < 16; ++i) {
    const int c = lane + (i << 5);
    x[i] = hp[c] + ap[c];
    s += x[i];
  }
#pragma unroll
  for (int off = 16; off > 0; off >>= 1) s += __shfl_xor(s, off, 32);
  const float mean = s * (1.0f / 512.0f);

  float vs = 0.f;
#pragma unroll
  for (int i = 0; i < 16; ++i) { const float d = x[i] - mean; vs += d * d; }
#pragma unroll
  for (int off = 16; off > 0; off >>= 1) vs += __shfl_xor(vs, off, 32);
  const float inv = rsqrtf(vs * (1.0f / 512.0f) + 1e-5f);

#pragma unroll
  for (int i = 0; i < 16; ++i) {
    const int c = lane + (i << 5);
    hp[c] = (x[i] - mean) * inv * gw[c] + bw[c];
  }
}

// h = h*sqrt(512) + PE(l, c), flat over 4096*512 elements
__global__ void pe_scale_kernel(float* __restrict__ h)
{
  const int idx = blockIdx.x * blockDim.x + threadIdx.x;
  const int c = idx & 511;
  const int t = idx >> 9;
  const int l = t & 1023;
  const float i2 = (float)(c & ~1);
  const float freq = __expf(i2 * (-9.210340371976184f / 512.0f)); // -ln(1e4)/512
  const float ang = (float)l * freq;
  const float pe = (c & 1) ? __cosf(ang) : __sinf(ang);
  h[idx] = h[idx] * 22.627416997969522f + pe; // sqrt(512)
}

extern "C" void kernel_launch(void* const* d_in, const int* in_sizes, int n_in,
                              void* d_out, int out_size, void* d_ws, size_t ws_size,
                              hipStream_t stream) {
  (void)in_sizes; (void)n_in; (void)out_size; (void)ws_size;
  const float* x     = (const float*)d_in[0];
  const float* W_emb = (const float*)d_in[1];
  const float* b_emb = (const float*)d_in[2];
  const float* Wq    = (const float*)d_in[3];
  const float* bq    = (const float*)d_in[4];
  const float* Wk    = (const float*)d_in[5];
  const float* bk    = (const float*)d_in[6];
  const float* Wv    = (const float*)d_in[7];
  const float* bv    = (const float*)d_in[8];
  const float* Wo    = (const float*)d_in[9];
  const float* bo    = (const float*)d_in[10];
  const float* ln1g  = (const float*)d_in[11];
  const float* ln1b  = (const float*)d_in[12];
  const float* W1    = (const float*)d_in[13];
  const float* b1    = (const float*)d_in[14];
  const float* W2    = (const float*)d_in[15];
  const float* b2    = (const float*)d_in[16];
  const float* ln2g  = (const float*)d_in[17];
  const float* ln2b  = (const float*)d_in[18];
  const float* W_dec = (const float*)d_in[19];
  const float* b_dec = (const float*)d_in[20];
  float* out = (float*)d_out;

  char* ws = (char*)d_ws;
  const size_t MB = 1024 * 1024;
  float* h      = (float*)(ws);              //  8 MB [4096,512]
  float* abuf   = (float*)(ws + 8  * MB);    //  8 MB [4096,512]
  float* q      = (float*)(ws + 16 * MB);    //  8 MB [4096,512]
  float* kbuf   = (float*)(ws + 24 * MB);    //  8 MB [4096,512]
  float* vbuf   = (float*)(ws + 32 * MB);    //  8 MB [4096,512]
  float* ffn    = (float*)(ws + 40 * MB);    // 32 MB [4096,2048]
  float* scores = (float*)(ws + 72 * MB);    // 128 MB [4,8,1024,1024]

  const dim3 blk(128);
  const long long Z0 = 0;

  auto gemm = [&](const float* A, const float* B, float* C, const float* bias,
                  int M, int N, int K, int lda, int ldb, int ldc,
                  float alpha, bool transB, bool relu, int batches,
                  long long aSB, long long aSH, long long bSB, long long bSH,
                  long long cSB, long long cSH) {
    dim3 grid(M / BM, N / BN, batches);
    if (transB)
      gemm_wmma_f16<true,  false><<<grid, blk, 0, stream>>>(A, B, C, bias, K, lda, ldb, ldc,
                                                            aSB, aSH, bSB, bSH, cSB, cSH, alpha);
    else if (relu)
      gemm_wmma_f16<false, true ><<<grid, blk, 0, stream>>>(A, B, C, bias, K, lda, ldb, ldc,
                                                            aSB, aSH, bSB, bSH, cSB, cSH, alpha);
    else
      gemm_wmma_f16<false, false><<<grid, blk, 0, stream>>>(A, B, C, bias, K, lda, ldb, ldc,
                                                            aSB, aSH, bSB, bSH, cSB, cSH, alpha);
  };

  // ---- embedding + positional encoding
  gemm(x, W_emb, h, b_emb, 4096, 512, 128, 128, 512, 512, 1.0f, false, false, 1,
       Z0, Z0, Z0, Z0, Z0, Z0);
  pe_scale_kernel<<<(4096 * 512) / 256, 256, 0, stream>>>(h);

  for (int L = 0; L < 3; ++L) {
    const long long wo = (long long)L * 512 * 512;
    // QKV projections
    gemm(h, Wq + wo, q,    bq + L * 512, 4096, 512, 512, 512, 512, 512, 1.0f, false, false, 1, Z0,Z0,Z0,Z0,Z0,Z0);
    gemm(h, Wk + wo, kbuf, bk + L * 512, 4096, 512, 512, 512, 512, 512, 1.0f, false, false, 1, Z0,Z0,Z0,Z0,Z0,Z0);
    gemm(h, Wv + wo, vbuf, bv + L * 512, 4096, 512, 512, 512, 512, 512, 1.0f, false, false, 1, Z0,Z0,Z0,Z0,Z0,Z0);
    // scores[b,h] = Q K^T / sqrt(64), batched over 32 (b,h)
    gemm(q, kbuf, scores, nullptr, 1024, 1024, 64, 512, 512, 1024, 0.125f, true, false, 32,
         (long long)1024 * 512, 64, (long long)1024 * 512, 64,
         (long long)8 * 1024 * 1024, (long long)1024 * 1024);
    // ProbSparse top-34 mask + softmax, in place
    topk_softmax_kernel<<<(32 * 1024) / 8, 256, 0, stream>>>(scores, 34);
    // head outputs (concat layout) = P @ V  -> abuf
    gemm(scores, vbuf, abuf, nullptr, 1024, 64, 1024, 1024, 512, 512, 1.0f, false, false, 32,
         (long long)8 * 1024 * 1024, (long long)1024 * 1024,
         (long long)1024 * 512, 64, (long long)1024 * 512, 64);
    // output projection -> q (reused), then h = LN(h + q)
    gemm(abuf, Wo + wo, q, bo + L * 512, 4096, 512, 512, 512, 512, 512, 1.0f, false, false, 1, Z0,Z0,Z0,Z0,Z0,Z0);
    add_layernorm_kernel<<<512, 256, 0, stream>>>(h, q, ln1g + L * 512, ln1b + L * 512);
    // FFN
    gemm(h, W1 + (long long)L * 512 * 2048, ffn, b1 + L * 2048,
         4096, 2048, 512, 512, 2048, 2048, 1.0f, false, true, 1, Z0,Z0,Z0,Z0,Z0,Z0);
    gemm(ffn, W2 + (long long)L * 2048 * 512, abuf, b2 + L * 512,
         4096, 512, 2048, 2048, 512, 512, 1.0f, false, false, 1, Z0,Z0,Z0,Z0,Z0,Z0);
    add_layernorm_kernel<<<512, 256, 0, stream>>>(h, abuf, ln2g + L * 512, ln2b + L * 512);
  }

  // ---- decoder
  gemm(h, W_dec, out, b_dec, 4096, 64, 512, 512, 64, 64, 1.0f, false, false, 1, Z0,Z0,Z0,Z0,Z0,Z0);
}
